// InstanceLoss_70738111365295
// MI455X (gfx1250) — matrix operand order repaired and checked by reading, not compiled, not run on hardware
//
#include <hip/hip_runtime.h>
#include <hip/hip_bf16.h>
#include <math.h>

// Problem dims (fixed by reference setup_inputs)
#define BB 8
#define MM 100
#define MT 8              // m tiles of 16 (padded 100 -> 128)
#define MPAD 128
#define NN 16384
#define DD 256
#define DT 16             // 256/16 d tiles
#define KS 512            // 16384/32 k-steps of 32

typedef __attribute__((ext_vector_type(16))) _Float16 v16h;
typedef __attribute__((ext_vector_type(8)))  float    v8f;

// ---- workspace layout ----
// Apack: BB*MT*KS frags * 1KB = 33,554,432 B ; Bpack: BB*DT*KS frags * 1KB = 67,108,864 B
static constexpr size_t OFF_APACK = 0;
static constexpr size_t OFF_BPACK = 33554432;
static constexpr size_t OFF_FSUM  = OFF_BPACK + 67108864;            // BB*DD f32
static constexpr size_t OFF_MSUM  = OFF_FSUM + (size_t)BB * DD * 4;  // BB*MM f32 (pad 4KB)
static constexpr size_t OFF_CENT  = OFF_MSUM + 4096;                 // BB*MPAD*DD f32
static constexpr size_t OFF_ACC   = OFF_CENT + (size_t)BB * MPAD * DD * 4;
// fragment index of first B fragment, in 1KB-fragment units (A frags = BB*MT*KS)
static constexpr size_t BFRAG0 = (size_t)BB * MT * KS;

__device__ __forceinline__ float wave_sum(float v) {
  #pragma unroll
  for (int off = 16; off > 0; off >>= 1) v += __shfl_xor(v, off, 32);
  return v;
}
__device__ __forceinline__ float sigmoidf(float x) {
  return 1.0f / (1.0f + __expf(-x));
}

// gfx1250 async global->LDS copy (ASYNCcnt), bypasses VGPRs entirely.
__device__ __forceinline__ void async_load_b128(unsigned int lds_off, const void* gaddr) {
  asm volatile("global_load_async_to_lds_b128 %0, %1, off"
               :: "v"(lds_off), "v"(gaddr) : "memory");
}
__device__ __forceinline__ void wait_asynccnt0() {
  asm volatile("s_wait_asynccnt 0" ::: "memory");
}

// zero atomically-accumulated buffers (must run every call: ws is not re-poisoned)
__global__ void il_init(float* fsum, float* msum, float* acc) {
  int t = threadIdx.x;
  for (int i = t; i < BB * DD; i += 256) fsum[i] = 0.0f;
  for (int i = t; i < BB * MM; i += 256) msum[i] = 0.0f;
  if (t < 2) acc[t] = 0.0f;
}

// ONE pass over features: row L2 norms + F_sum[b,d] + B-fragment packing.
// block = (b, ks): 32 rows x 256 cols staged in LDS. grid BB*KS x 256
__global__ void il_featpack(const float* __restrict__ feats,
                            _Float16* __restrict__ bpack, float* __restrict__ fsum) {
  __shared__ float tile[32][DD + 1];
  __shared__ float invr[32];
  int ks = blockIdx.x & (KS - 1);
  int b  = blockIdx.x >> 9;
  int t = threadIdx.x, wid = t >> 5, lane = t & 31;
  int n0 = ks * 32;
  #pragma unroll 4
  for (int i = 0; i < 32; ++i)
    tile[i][t] = feats[((size_t)b * NN + n0 + i) * DD + t];
  __syncthreads();
  #pragma unroll
  for (int r4 = 0; r4 < 4; ++r4) {
    int r = wid * 4 + r4;
    float s = 0.0f;
    #pragma unroll
    for (int j = 0; j < 8; ++j) { float v = tile[r][lane * 8 + j]; s += v * v; }
    s = wave_sum(s);
    if (lane == 0) invr[r] = 1.0f / fmaxf(sqrtf(s), 1e-12f);
  }
  __syncthreads();
  float fs = 0.0f;
  #pragma unroll 4
  for (int i = 0; i < 32; ++i) fs += tile[i][t] * invr[i];
  atomicAdd(&fsum[b * DD + t], fs);
  // WMMA B-fragment (32x16 f16): lane l -> col N=l%16, K = (h<8?h:h+8) + (l>=16?8:0)
  #pragma unroll 2
  for (int q = 0; q < 32; ++q) {
    int eg = t + 256 * q;
    int dt = eg >> 9;
    int e  = eg & 511;
    int fl = e >> 4, h = e & 15;
    int n_loc = ((h < 8) ? h : h + 8) + ((fl >= 16) ? 8 : 0);
    int d_loc = fl & 15;
    float v = tile[n_loc][dt * 16 + d_loc] * invr[n_loc];
    bpack[((size_t)(b * DT + dt) * KS + ks) * 512 + e] = (_Float16)v;
  }
}

// ONE pass over pred_masks: sigmoid + row sums + A-fragment packing.
// block = (b, mt, kc of 32 ksteps). grid BB*MT*16 x 256
__global__ void il_maskpack(const float* __restrict__ masks,
                            _Float16* __restrict__ apack, float* __restrict__ msum) {
  int g = blockIdx.x;
  int kc = g & 15; g >>= 4;
  int mt = g & 7;  g >>= 3;
  int b  = g;
  int t = threadIdx.x;
  int r = t >> 4;
  float rowsum = 0.0f;
  for (int q = 0; q < 32; ++q) {
    int ks = kc * 32 + q;
    size_t fbase = ((size_t)(b * MT + mt) * KS + ks) * 512;
    #pragma unroll
    for (int p = 0; p < 2; ++p) {
      int e = t + p * 256;
      int fl = e >> 4, h = e & 15;
      int k_loc = ((h < 8) ? h : h + 8) + ((fl >= 16) ? 8 : 0);
      int m = mt * 16 + (fl & 15);
      float v = 0.0f;
      if (m < MM)
        v = sigmoidf(masks[((size_t)b * MM + m) * NN + ks * 32 + k_loc]);
      apack[fbase + e] = (_Float16)v;
      rowsum += v;
    }
  }
  #pragma unroll
  for (int off = 8; off > 0; off >>= 1) rowsum += __shfl_xor(rowsum, off, 32);
  int m = mt * 16 + r;
  if ((t & 15) == 0 && m < MM) atomicAdd(&msum[b * MM + m], rowsum);
}

// centroid_raw GEMM: 8 waves cover 2 m-tiles x 4 d-tiles. Stages of 4 ksteps
// (24KB) double-buffered via async global->LDS; running src pointers, no selects.
// Inner loop: ds_load_b128 -> v_wmma_f32_16x16x32_f16. grid BB*4*4 x 256
__global__ void il_gemm(const uint4* __restrict__ wsu4, float* __restrict__ cent) {
  __shared__ __align__(128) uint4 buf[2][2048];   // 2 x 32KB (24KB used per stage)
  int bx = blockIdx.x;
  int dtg = bx & 3; bx >>= 2;
  int mtg = bx & 3; bx >>= 2;
  int b   = bx;
  int t = threadIdx.x, wid = t >> 5, lane = t & 31;
  int mt_local = wid >> 2, dt_local = wid & 3;
  int mt = mtg * 2 + mt_local;
  int dt = dtg * 4 + dt_local;

  // copy plan: 24 chunks of 1KB per stage (chunk = id*4 + j; id 0-1 A, 2-5 B),
  // 6 async b128 transfers per thread per stage.
  const uint4* src[6];
  unsigned int lds0[6];
  unsigned int bufbase = (unsigned int)(uintptr_t)&buf[0][0];
  #pragma unroll
  for (int i = 0; i < 6; ++i) {
    int u = t + 256 * i;                 // 0..1535
    int chunk = u >> 6, within = u & 63;
    int j = chunk & 3, id = chunk >> 2;
    size_t frag = (id < 2) ? ((size_t)(b * MT + mtg * 2 + id) * KS)
                           : (BFRAG0 + (size_t)(b * DT + dtg * 4 + (id - 2)) * KS);
    src[i]  = wsu4 + (frag + j) * 64 + within;
    lds0[i] = bufbase + (unsigned int)(chunk * 1024 + within * 16);
  }

  // preload stage 0 (ksteps 0..3)
  #pragma unroll
  for (int i = 0; i < 6; ++i) {
    async_load_b128(lds0[i], src[i]);
    src[i] += 256;                       // advance 4 ksteps
  }
  wait_asynccnt0();
  __syncthreads();

  v8f acc = {};
  for (int s = 0; s < 128; ++s) {
    if (s + 1 < 128) {
      unsigned int boff = (unsigned int)((s + 1) & 1) * 32768u;
      #pragma unroll
      for (int i = 0; i < 6; ++i) {
        async_load_b128(lds0[i] + boff, src[i]);
        src[i] += 256;
      }
    }
    const v16h* base = ((const v16h*)buf) + (s & 1) * 1024;   // 32 v16h per chunk
    #pragma unroll
    for (int j = 0; j < 4; ++j) {
      v16h a  = base[(mt_local * 4 + j) * 32 + lane];
      v16h bb = base[((2 + dt_local) * 4 + j) * 32 + lane];
      acc = __builtin_amdgcn_wmma_f32_16x16x32_f16(false, a, false, bb,
                                                   (short)0, acc, false, false);
    }
    wait_asynccnt0();      // this wave's next-stage transfers landed in LDS
    __syncthreads();       // publish to the other waves
  }

  // C layout: VGPR r, lanes 0-15 -> M=r, lanes 16-31 -> M=r+8; N = lane%16
  int row_off = (lane >= 16) ? 8 : 0;
  int col = lane & 15;
  float* cbase = cent + ((size_t)(b * MPAD + mt * 16) * DD + dt * 16);
  #pragma unroll
  for (int r = 0; r < 8; ++r)
    cbase[(size_t)(r + row_off) * DD + col] = acc[r];
}

// per-instance: intra = ||c||/mask_sum ; inter = (Fsum.c_norm - ||c||)/anti_sum ; hinge
__global__ void il_finalize(const float* __restrict__ cent, const float* __restrict__ fsum,
                            const float* __restrict__ msum, const float* __restrict__ scores,
                            float* __restrict__ acc) {
  int wid = threadIdx.x >> 5, lane = threadIdx.x & 31;
  int inst = blockIdx.x * 8 + wid;
  if (inst >= BB * MM) return;
  int b = inst / MM, m = inst % MM;
  const float* c = cent + (size_t)(b * MPAD + m) * DD;
  const float* F = fsum + b * DD;
  float s2 = 0.0f, dF = 0.0f;
  #pragma unroll
  for (int j = 0; j < 8; ++j) {
    int d = lane * 8 + j;
    float cv = c[d];
    s2 += cv * cv;
    dF += cv * F[d];
  }
  s2 = wave_sum(s2);
  dF = wave_sum(dF);
  if (lane == 0) {
    float norm = sqrtf(s2);
    float inv = 1.0f / fmaxf(norm, 1e-12f);
    float ms = msum[inst];
    float intra = (s2 * inv) / (ms + 1e-8f);
    float inter = (dF * inv - s2 * inv) / ((float)NN - ms + 1e-8f);
    float gap = fmaxf(inter - intra + 0.3f, 0.0f);
    float sc = scores[inst];
    if (sc >= 0.1f) {
      atomicAdd(&acc[0], sc * gap);
      atomicAdd(&acc[1], 1.0f);
    }
  }
}

__global__ void il_writeout(const float* __restrict__ acc, float* __restrict__ out) {
  out[0] = 0.3f * acc[0] / fmaxf(acc[1], 1.0f);
}

extern "C" void kernel_launch(void* const* d_in, const int* in_sizes, int n_in,
                              void* d_out, int out_size, void* d_ws, size_t ws_size,
                              hipStream_t stream) {
  const float* pred_masks  = (const float*)d_in[0];   // (8,100,16384)
  const float* pred_scores = (const float*)d_in[1];   // (8,100)
  const float* features    = (const float*)d_in[2];   // (8,16384,256)
  float* out = (float*)d_out;

  char* ws = (char*)d_ws;
  _Float16* apack = (_Float16*)(ws + OFF_APACK);
  _Float16* bpack = (_Float16*)(ws + OFF_BPACK);
  float* fsum = (float*)(ws + OFF_FSUM);
  float* msum = (float*)(ws + OFF_MSUM);
  float* cent = (float*)(ws + OFF_CENT);
  float* acc  = (float*)(ws + OFF_ACC);

  il_init<<<1, 256, 0, stream>>>(fsum, msum, acc);
  il_featpack<<<BB * KS, 256, 0, stream>>>(features, bpack, fsum);
  il_maskpack<<<BB * MT * 16, 256, 0, stream>>>(pred_masks, apack, msum);
  il_gemm<<<BB * 16, 256, 0, stream>>>((const uint4*)ws, cent);
  il_finalize<<<100, 256, 0, stream>>>(cent, fsum, msum, pred_scores, acc);
  il_writeout<<<1, 1, 0, stream>>>(acc, out);
}
